// UnetTransformer_8117488190283
// MI455X (gfx1250) — matrix operand order repaired and checked by reading, not compiled, not run on hardware
//
#include <hip/hip_runtime.h>
#include <math.h>

// ---------------------------------------------------------------------------
// Types for CDNA5 WMMA (wave32): v_wmma_f32_16x16x32_bf16
// ---------------------------------------------------------------------------
typedef __attribute__((ext_vector_type(16))) __bf16 v16bf;
typedef __attribute__((ext_vector_type(8)))  __bf16 v8bf;   // 16 bytes
typedef __attribute__((ext_vector_type(4)))  __bf16 v4bf;   // 8 bytes
typedef __attribute__((ext_vector_type(2)))  __bf16 v2bf;   // 4 bytes
typedef __attribute__((ext_vector_type(8)))  float  v8f;

__device__ __forceinline__ v8f wmma_bf16(v16bf a, v16bf b, v8f c) {
  return __builtin_amdgcn_wmma_f32_16x16x32_bf16(false, a, false, b, (short)0, c,
                                                 false, false);
}

__device__ __forceinline__ v8bf cvt8_bf16(float4 a, float4 b) {
  v8bf d;
  d[0] = (__bf16)a.x; d[1] = (__bf16)a.y; d[2] = (__bf16)a.z; d[3] = (__bf16)a.w;
  d[4] = (__bf16)b.x; d[5] = (__bf16)b.y; d[6] = (__bf16)b.z; d[7] = (__bf16)b.w;
  return d;
}

// ---------------------------------------------------------------------------
// GEMM: C[M,N] = A[M,K] @ W[K,N] (+bias) (+relu), fp32 in/out, bf16 WMMA.
// Block tile 128x128x32, 256 threads = 8 waves (2x4), wave tile 64x32.
// REQUIRES: M % 128 == 0, N % 128 == 0, K % 32 == 0 (true for all call sites;
// guards removed so staging stays branch-free and vectorized).
// flags: 1 = bias, 2 = relu
// ---------------------------------------------------------------------------
__global__ __launch_bounds__(256) void gemm_bf16_k(
    const float* __restrict__ A, const float* __restrict__ W,
    const float* __restrict__ bias, float* __restrict__ C, int M, int N, int K,
    int flags) {
  __shared__ __bf16 As[128][40];  // stride 40 -> 80 B rows (16B-aligned)
  __shared__ __bf16 Bs[128][40];  // transposed: Bs[n][k]

  const int tid  = threadIdx.x;
  const int wave = tid >> 5;
  const int lane = tid & 31;
  const int lh   = lane >> 4;
  const int l16  = lane & 15;

  const int bm = blockIdx.y * 128;
  const int bn = blockIdx.x * 128;
  const int wm = (wave >> 2) * 64;
  const int wn = (wave & 3) * 32;

  // staging addresses (advance by K-step each iteration)
  const int ar = tid >> 1;            // A row within tile
  const int ac = (tid & 1) * 16;      // A col group (0 or 16)
  const float* aSrc = A + (size_t)(bm + ar) * K + ac;
  const int bn0 = (tid & 31) * 4;     // B: 4 n-values
  const int bk0 = (tid >> 5) * 4;     // B: 4 k-values
  const float* bSrc = W + (size_t)bk0 * N + bn + bn0;

  v8f acc[4][2];
  const v8f vzero = {};
#pragma unroll
  for (int i = 0; i < 4; ++i)
#pragma unroll
    for (int j = 0; j < 2; ++j) acc[i][j] = vzero;

  for (int k0 = 0; k0 < K; k0 += 32) {
    // ---- stage A tile (128x32): 16 consecutive floats -> 2x b128 stores ----
    {
      float4 f0 = *(const float4*)(aSrc);
      float4 f1 = *(const float4*)(aSrc + 4);
      float4 f2 = *(const float4*)(aSrc + 8);
      float4 f3 = *(const float4*)(aSrc + 12);
      *(v8bf*)&As[ar][ac]     = cvt8_bf16(f0, f1);
      *(v8bf*)&As[ar][ac + 8] = cvt8_bf16(f2, f3);
      if (k0 + 32 < K) __builtin_prefetch(aSrc + 32, 0, 1);
      aSrc += 32;
    }
    // ---- stage B tile (32x128) transposed: 4x4 micro-transpose, b64 stores --
    {
      float4 r0 = *(const float4*)(bSrc);
      float4 r1 = *(const float4*)(bSrc + (size_t)N);
      float4 r2 = *(const float4*)(bSrc + 2 * (size_t)N);
      float4 r3 = *(const float4*)(bSrc + 3 * (size_t)N);
      const float* c0 = (const float*)&r0;
      const float* c1 = (const float*)&r1;
      const float* c2 = (const float*)&r2;
      const float* c3 = (const float*)&r3;
#pragma unroll
      for (int n = 0; n < 4; ++n) {
        v4bf q;
        q[0] = (__bf16)c0[n]; q[1] = (__bf16)c1[n];
        q[2] = (__bf16)c2[n]; q[3] = (__bf16)c3[n];
        *(v4bf*)&Bs[bn0 + n][bk0] = q;
      }
      if (k0 + 32 < K) __builtin_prefetch(bSrc + 32 * (size_t)N, 0, 1);
      bSrc += 32 * (size_t)N;
    }
    __syncthreads();

    // ---- fragments (ISA VGPR layouts) ----
    v16bf afr[4], bfr[2];
#pragma unroll
    for (int i = 0; i < 4; ++i) {
      const __bf16* p = &As[wm + i * 16 + l16][lh * 8];
      ((uint4*)&afr[i])[0] = *(const uint4*)p;         // k = lh*8 .. +7
      ((uint4*)&afr[i])[1] = *(const uint4*)(p + 16);  // k = 16+lh*8 .. +7
    }
#pragma unroll
    for (int j = 0; j < 2; ++j) {
      const __bf16* p = &Bs[wn + j * 16 + l16][lh * 8];
      ((uint4*)&bfr[j])[0] = *(const uint4*)p;         // k = lh*8 .. +7
      ((uint4*)&bfr[j])[1] = *(const uint4*)(p + 16);  // k = 16+lh*8 .. +7
    }
#pragma unroll
    for (int i = 0; i < 4; ++i)
#pragma unroll
      for (int j = 0; j < 2; ++j)
        acc[i][j] = wmma_bf16(afr[i], bfr[j], acc[i][j]);
    __syncthreads();
  }

  // ---- epilogue (no guards: exact tiling) ----
#pragma unroll
  for (int j = 0; j < 2; ++j) {
    const int col = bn + wn + j * 16 + l16;
    const float bj = (flags & 1) ? bias[col] : 0.0f;
#pragma unroll
    for (int i = 0; i < 4; ++i) {
      const int row = bm + wm + i * 16 + lh * 8;
#pragma unroll
      for (int r = 0; r < 8; ++r) {
        float v = acc[i][j][r] + bj;
        if (flags & 2) v = fmaxf(v, 0.0f);
        C[(size_t)(row + r) * N + col] = v;
      }
    }
  }
}

// ---------------------------------------------------------------------------
// Flash attention (dh=64), bf16 WMMA for QK^T and PV, online softmax.
// Block = 128 threads = 4 waves; each wave owns 16 query rows; all waves share
// one (b,h), so K/V chunks are staged cooperatively in LDS once per block.
// ---------------------------------------------------------------------------
__global__ __launch_bounds__(128) void attn_wmma_k(
    const float* __restrict__ Qb, const float* __restrict__ Kb,
    const float* __restrict__ Vb, float* __restrict__ Ob, int S_, int H_,
    int qStride, int qHead, int qOff, int kStride, int kHead, int kOff,
    int vStride, int vHead, int vOff, int oStride, int oHead, int causal) {
  __shared__ __bf16 Ks[32][72];      // [key][dh]  (stride 144 B, 16B-aligned)
  __shared__ __bf16 Vt[64][40];      // [dh][key]  (stride 80 B)
  __shared__ __bf16 Pl[4][16][40];   // per-wave P staging (C- -> A-layout)

  const int tid  = threadIdx.x;
  const int w    = tid >> 5;
  const int lane = tid & 31;
  const int lh   = lane >> 4;
  const int l16  = lane & 15;
  const int b    = blockIdx.y / H_;
  const int h    = blockIdx.y % H_;
  const int q0   = blockIdx.x * 64 + w * 16;

  const float* Q = Qb + (size_t)b * S_ * qStride + (size_t)h * qHead + qOff;
  const float* K = Kb + (size_t)b * S_ * kStride + (size_t)h * kHead + kOff;
  const float* V = Vb + (size_t)b * S_ * vStride + (size_t)h * vHead + vOff;
  float*       O = Ob + (size_t)b * S_ * oStride + (size_t)h * oHead;

  // Q fragments (scale 1/8 = rsqrt(64) folded in), two dh-chunks of 32
  v16bf qf[2];
  {
    const float* qp = Q + (size_t)(q0 + l16) * qStride;
#pragma unroll
    for (int c = 0; c < 2; ++c) {
      const float* r0 = qp + c * 32 + lh * 8;
      const float* r1 = r0 + 16;
#pragma unroll
      for (int i = 0; i < 8; ++i) qf[c][i]     = (__bf16)(r0[i] * 0.125f);
#pragma unroll
      for (int i = 0; i < 8; ++i) qf[c][8 + i] = (__bf16)(r1[i] * 0.125f);
    }
  }

  float mrow[8], lrow[8];
  v8f oacc[4];
  const v8f vzero = {};
#pragma unroll
  for (int r = 0; r < 8; ++r) { mrow[r] = -3.0e38f; lrow[r] = 0.0f; }
#pragma unroll
  for (int t = 0; t < 4; ++t) oacc[t] = vzero;

  // per-wave chunk count; block-uniform loop bound so barriers stay uniform
  const int my_nkt  = causal ? (((q0 + 15) >> 5) + 1) : (S_ >> 5);
  const int blk_nkt = causal ? ((blockIdx.x * 64 + 63) >> 5) + 1 : (S_ >> 5);

  // staging indices (128 threads)
  const int skr = tid >> 2;            // K: key row 0..31
  const int skc = (tid & 3) * 16;      // K: dh col group
  const int svk = (tid & 15) * 2;      // V: key pair
  const int svd = (tid >> 4) * 8;      // V: dh group of 8

  for (int kt = 0; kt < blk_nkt; ++kt) {
    const int kbase = kt * 32;
    // ---- cooperative staging: K chunk row-major, V chunk transposed ----
    {
      const float* kp = K + (size_t)(kbase + skr) * kStride + skc;
      float4 f0 = *(const float4*)(kp);
      float4 f1 = *(const float4*)(kp + 4);
      float4 f2 = *(const float4*)(kp + 8);
      float4 f3 = *(const float4*)(kp + 12);
      *(v8bf*)&Ks[skr][skc]     = cvt8_bf16(f0, f1);
      *(v8bf*)&Ks[skr][skc + 8] = cvt8_bf16(f2, f3);

      const float* vp0 = V + (size_t)(kbase + svk) * vStride + svd;
      const float* vp1 = vp0 + vStride;
      float4 a0 = *(const float4*)(vp0);
      float4 a1 = *(const float4*)(vp0 + 4);
      float4 b0 = *(const float4*)(vp1);
      float4 b1 = *(const float4*)(vp1 + 4);
      const float* x0 = (const float*)&a0;
      const float* x1 = (const float*)&a1;
      const float* y0 = (const float*)&b0;
      const float* y1 = (const float*)&b1;
#pragma unroll
      for (int d = 0; d < 4; ++d) {
        v2bf p0; p0[0] = (__bf16)x0[d]; p0[1] = (__bf16)y0[d];
        *(v2bf*)&Vt[svd + d][svk] = p0;
        v2bf p1; p1[0] = (__bf16)x1[d]; p1[1] = (__bf16)y1[d];
        *(v2bf*)&Vt[svd + 4 + d][svk] = p1;
      }
    }
    __syncthreads();

    if (kt < my_nkt) {
      // ---- scores: two 16-key sub-tiles ----
      v8f s[2];
#pragma unroll
      for (int sub = 0; sub < 2; ++sub) {
        const int k16 = kbase + sub * 16;
        v8f a = vzero;
        if (!(causal && k16 > q0 + 15)) {
          const __bf16* kp = &Ks[sub * 16 + l16][lh * 8];
          v16bf kfA, kfB;
          ((uint4*)&kfA)[0] = *(const uint4*)kp;          // dh = lh*8 .., 16+lh*8 ..
          ((uint4*)&kfA)[1] = *(const uint4*)(kp + 16);
          ((uint4*)&kfB)[0] = *(const uint4*)(kp + 32);   // dh = 32+lh*8 .., 48+lh*8 ..
          ((uint4*)&kfB)[1] = *(const uint4*)(kp + 48);
          a = wmma_bf16(qf[0], kfA, a);
          a = wmma_bf16(qf[1], kfB, a);
          if (causal) {
#pragma unroll
            for (int r = 0; r < 8; ++r)
              if (k16 + l16 > q0 + lh * 8 + r) a[r] = -9.0e15f;
          }
        } else {
#pragma unroll
          for (int r = 0; r < 8; ++r) a[r] = -9.0e15f;
        }
        s[sub] = a;
      }

      // ---- joint online softmax over 32-key chunk ----
#pragma unroll
      for (int r = 0; r < 8; ++r) {
        float v0 = s[0][r], v1 = s[1][r];
        float tm = fmaxf(v0, v1);
#pragma unroll
        for (int o = 1; o < 16; o <<= 1) tm = fmaxf(tm, __shfl_xor(tm, o, 32));
        const float mnew = fmaxf(mrow[r], tm);
        const float p0 = __expf(v0 - mnew);
        const float p1 = __expf(v1 - mnew);
        float ps = p0 + p1;
#pragma unroll
        for (int o = 1; o < 16; o <<= 1) ps += __shfl_xor(ps, o, 32);
        const float corr = __expf(mrow[r] - mnew);
        lrow[r] = lrow[r] * corr + ps;
        mrow[r] = mnew;
#pragma unroll
        for (int t = 0; t < 4; ++t) oacc[t][r] *= corr;
        Pl[w][lh * 8 + r][l16]      = (__bf16)p0;
        Pl[w][lh * 8 + r][l16 + 16] = (__bf16)p1;
      }

      // ---- P as A-fragment; PV over 4 dh n-tiles ----
      v16bf pf;
      {
        const __bf16* pp = &Pl[w][l16][lh * 8];
        ((uint4*)&pf)[0] = *(const uint4*)pp;
        ((uint4*)&pf)[1] = *(const uint4*)(pp + 16);
      }
#pragma unroll
      for (int t = 0; t < 4; ++t) {
        const __bf16* vp = &Vt[t * 16 + l16][lh * 8];
        v16bf vf;
        ((uint4*)&vf)[0] = *(const uint4*)vp;
        ((uint4*)&vf)[1] = *(const uint4*)(vp + 16);
        oacc[t] = wmma_bf16(pf, vf, oacc[t]);
      }
    }
    __syncthreads();
  }

  // ---- normalize + store merged output ----
#pragma unroll
  for (int r = 0; r < 8; ++r) {
    const float inv = 1.0f / lrow[r];
    const int row = q0 + lh * 8 + r;
#pragma unroll
    for (int t = 0; t < 4; ++t)
      O[(size_t)row * oStride + t * 16 + l16] = oacc[t][r] * inv;
  }
}

// ---------------------------------------------------------------------------
// LayerNorm: Out = LN(A (+R)) * g + b, optional ReLU. One block per row.
// flags: 1=add residual R, 2=relu after
// ---------------------------------------------------------------------------
__global__ __launch_bounds__(256) void layernorm_k(
    const float* __restrict__ A, const float* __restrict__ R,
    const float* __restrict__ g, const float* __restrict__ bt,
    float* __restrict__ Out, int Dd, int flags) {
  __shared__ float red[16];
  const int row = blockIdx.x;
  const int tid = threadIdx.x;
  const float* a  = A + (size_t)row * Dd;
  const float* rr = R + (size_t)row * Dd;  // only dereferenced if flags&1

  float s1 = 0.0f, s2 = 0.0f;
  for (int c = tid; c < Dd; c += blockDim.x) {
    float v = a[c];
    if (flags & 1) v += rr[c];
    s1 += v;
    s2 += v * v;
  }
#pragma unroll
  for (int o = 1; o < 32; o <<= 1) {
    s1 += __shfl_xor(s1, o, 32);
    s2 += __shfl_xor(s2, o, 32);
  }
  const int wv = tid >> 5;
  if ((tid & 31) == 0) { red[wv] = s1; red[8 + wv] = s2; }
  __syncthreads();
  if (tid == 0) {
    float t1 = 0.0f, t2 = 0.0f;
    const int nw = blockDim.x >> 5;
    for (int i = 0; i < nw; ++i) { t1 += red[i]; t2 += red[8 + i]; }
    red[0] = t1; red[8] = t2;
  }
  __syncthreads();
  const float mean = red[0] / (float)Dd;
  const float var  = red[8] / (float)Dd - mean * mean;
  const float rstd = rsqrtf(var + 1e-5f);
  for (int c = tid; c < Dd; c += blockDim.x) {
    float v = a[c];
    if (flags & 1) v += rr[c];
    float y = (v - mean) * rstd * g[c] + bt[c];
    if (flags & 2) y = fmaxf(y, 0.0f);
    Out[(size_t)row * Dd + c] = y;
  }
}

// ---------------------------------------------------------------------------
// Embedding gather + sinusoidal positional encoding (computed analytically)
// ---------------------------------------------------------------------------
__global__ __launch_bounds__(256) void embed_pe_k(
    const int* __restrict__ tok, const float* __restrict__ emb,
    float* __restrict__ out, int S_, int Dd) {
  const int s = blockIdx.x;
  const int b = blockIdx.y;
  const int t = tok[b * S_ + s];
  const float nl = -9.210340371976184f / (float)Dd;  // -ln(10000)/D
  float* o = out + ((size_t)b * S_ + s) * Dd;
  const float* e = emb + (size_t)t * Dd;
  for (int c = threadIdx.x; c < Dd; c += blockDim.x) {
    const float freq = __expf((float)(c & ~1) * nl);
    const float ang  = (float)s * freq;
    const float pe   = (c & 1) ? __cosf(ang) : __sinf(ang);
    o[c] = e[c] + pe;
  }
}

// ---------------------------------------------------------------------------
// Host orchestration
// ---------------------------------------------------------------------------
extern "C" void kernel_launch(void* const* d_in, const int* in_sizes, int n_in,
                              void* d_out, int out_size, void* d_ws,
                              size_t ws_size, hipStream_t stream) {
  (void)in_sizes; (void)n_in; (void)out_size; (void)ws_size;
  const int Bc = 2, Sc = 1024, Dc = 1024, Hc = 16, Fc = 4096, NCc = 32000;
  const int Mrows = Bc * Sc;
  const size_t BSD = (size_t)Mrows * Dc;

  // Input leaf order: jax tree_leaves => dict keys sorted alphabetically,
  // tuples (w,b)/(g,b) in order.
  //   0: x   1: output_shifted
  //   2..49 : dec[0..2], 16 leaves each:
  //     +0 ff1.w +1 ff1.b +2 ff2.w +3 ff2.b +4 ln1.g +5 ln1.b +6 ln2.g
  //     +7 ln2.b +8 ln3.g +9 ln3.b +10 o_c.w +11 o_c.b +12 o_x.w +13 o_x.b
  //     +14 qkv.w +15 qkv.b
  //   50: emb_dec  51: emb_enc
  //   52..87: enc[0..2], 12 leaves each:
  //     +0 ff1.w +1 ff1.b +2 ff2.w +3 ff2.b +4 ln1.g +5 ln1.b +6 ln2.g
  //     +7 ln2.b +8 o.w +9 o.b +10 qkv.w +11 qkv.b
  //   88 out1.w 89 out1.b 90 out2.w 91 out2.b 92 out_ln.g 93 out_ln.b
  const int DEC0 = 2, EMB_DEC = 50, EMB_ENC = 51, ENC0 = 52;
  const int OUT1W = 88, OUT1B = 89, OUT2W = 90, OUT2B = 91, OLNG = 92,
            OLNB = 93;

  const int* x_tok = (const int*)d_in[0];
  const int* y_tok = (const int*)d_in[1];
  auto P = [&](int i) { return (const float*)d_in[i]; };

  float* f      = (float*)d_ws;
  float* xe     = f;             // encoder stack[0]
  float* e1     = f + 1 * BSD;   // enc block 0 out (stack[1])
  float* e2     = f + 2 * BSD;   // enc block 1 out (stack[2])
  float* ha     = f + 3 * BSD;   // decoder hidden ping
  float* hb     = f + 4 * BSD;   // decoder hidden pong
  float* t_attn = f + 5 * BSD;
  float* t_proj = f + 6 * BSD;
  float* t_x2   = f + 7 * BSD;
  float* t_x5   = f + 8 * BSD;
  float* t_qkv  = f + 9 * BSD;   // [M, 3D]
  float* t_ff   = f + 12 * BSD;  // [M, F]

  auto gemm = [&](const float* A, const float* W, const float* bias, float* C,
                  int M, int N, int K, int flags) {
    dim3 g((N + 127) / 128, (M + 127) / 128);
    gemm_bf16_k<<<g, 256, 0, stream>>>(A, W, bias, C, M, N, K, flags);
  };
  auto ln = [&](const float* A, const float* R, int gi, int bi, float* O,
                int flags) {
    layernorm_k<<<Mrows, 256, 0, stream>>>(A, R, P(gi), P(bi), O, Dc, flags);
  };
  auto attn = [&](const float* Qb, int qs, int qh, int qo, const float* Kb,
                  int ks, int kh, int ko, const float* Vb, int vs, int vh,
                  int vo, float* Ob, int causal) {
    attn_wmma_k<<<dim3(Sc / 64, Bc * Hc), 128, 0, stream>>>(
        Qb, Kb, Vb, Ob, Sc, Hc, qs, qh, qo, ks, kh, ko, vs, vh, vo, Dc, 64,
        causal);
  };

  // Embeddings + positional encoding
  embed_pe_k<<<dim3(Sc, Bc), 256, 0, stream>>>(x_tok, P(EMB_ENC), xe, Sc, Dc);
  embed_pe_k<<<dim3(Sc, Bc), 256, 0, stream>>>(y_tok, P(EMB_DEC), ha, Sc, Dc);

  // Encoder blocks (block 2 output is never consumed by decoders -> skipped)
  auto enc_block = [&](const float* in, float* out, int base) {
    gemm(in, P(base + 10), P(base + 11), t_qkv, Mrows, 3 * Dc, Dc, 1);
    attn(t_qkv, 3 * Dc, 192, 0, t_qkv, 3 * Dc, 192, 64, t_qkv, 3 * Dc, 192,
         128, t_attn, 0);
    gemm(t_attn, P(base + 8), P(base + 9), t_proj, Mrows, Dc, Dc, 1);
    ln(t_proj, in, base + 4, base + 5, t_x2, 1);
    gemm(t_x2, P(base + 0), P(base + 1), t_ff, Mrows, Fc, Dc, 1 | 2);
    gemm(t_ff, P(base + 2), P(base + 3), t_proj, Mrows, Dc, Fc, 1);
    ln(t_proj, t_x2, base + 6, base + 7, out, 1);
  };
  enc_block(xe, e1, ENC0);
  enc_block(e1, e2, ENC0 + 12);

  // Decoder blocks: self-attn (causal) -> cross-attn(skip) -> FFN
  auto dec_block = [&](const float* in, const float* skip, float* out,
                       int base) {
    gemm(in, P(base + 14), P(base + 15), t_qkv, Mrows, 3 * Dc, Dc, 1);
    attn(t_qkv, 3 * Dc, 192, 0, t_qkv, 3 * Dc, 192, 64, t_qkv, 3 * Dc, 192,
         128, t_attn, 1);
    gemm(t_attn, P(base + 10), P(base + 11), t_proj, Mrows, Dc, Dc, 1);
    ln(t_proj, in, base + 4, base + 5, t_x2, 1);
    attn(t_x2, Dc, 64, 0, skip, Dc, 64, 0, skip, Dc, 64, 0, t_attn, 0);
    gemm(t_attn, P(base + 12), P(base + 13), t_proj, Mrows, Dc, Dc, 1);
    ln(t_proj, t_x2, base + 6, base + 7, t_x5, 1);
    gemm(t_x5, P(base + 0), P(base + 1), t_ff, Mrows, Fc, Dc, 1 | 2);
    gemm(t_ff, P(base + 2), P(base + 3), t_proj, Mrows, Dc, Fc, 1);
    ln(t_proj, t_x5, base + 8, base + 9, out, 1);
  };
  dec_block(ha, xe, hb, DEC0);
  dec_block(hb, e1, ha, DEC0 + 16);
  dec_block(ha, e2, hb, DEC0 + 32);

  // Output head: Linear -> LN -> ReLU -> vocab Linear (the 134 GF GEMM;
  // 131 MB weight fits in the 192 MB L2, so HBM reads it once)
  gemm(hb, P(OUT1W), P(OUT1B), t_proj, Mrows, Dc, Dc, 1);
  ln(t_proj, nullptr, OLNG, OLNB, t_x2, 2);
  gemm(t_x2, P(OUT2W), P(OUT2B), (float*)d_out, Mrows, NCc, Dc, 1);
}
